// LSTM_59502476919269
// MI455X (gfx1250) — matrix operand order repaired and checked by reading, not compiled, not run on hardware
//
#include <hip/hip_runtime.h>

// LSTM, B=256, T=4096, I=28, H=128, OUT=10.
// Output depends only on batch row 255. LSTM rows are independent, so the
// 16-row WMMA tile is filled with 16 identical clones of row 255 (C-row init
// is broadcast). gx = x[255] @ w_ih^T + biases is precomputed in parallel;
// the serial scan then only does the h @ w_hh^T recurrence with WMMA f16.

typedef __attribute__((ext_vector_type(16))) _Float16 v16h;
typedef __attribute__((ext_vector_type(8)))  float    v8f;

#define BATCH   256
#define TSTEPS  4096
#define NIN     28
#define HDIM    128
#define G4      512
#define HSTRIDE 136   // f16 elems per LDS row; 272B stride -> no 64-bank conflicts

__device__ __forceinline__ float fastrcp(float x)   { return __builtin_amdgcn_rcpf(x); }
__device__ __forceinline__ float sigf(float x)      { return fastrcp(1.f + __expf(-x)); }
__device__ __forceinline__ float tanh_fast(float x) { return 1.f - 2.f * fastrcp(__expf(2.f * x) + 1.f); }

// ---------------------------------------------------------------------------
// Kernel 1: gx[t, g] = x[255, t, :] . w_ih[g, :] + b_ih[g] + b_hh[g]
// [4096 x 512], trivially parallel.
// ---------------------------------------------------------------------------
__global__ void gx_kernel(const float* __restrict__ x, const float* __restrict__ w_ih,
                          const float* __restrict__ b_ih, const float* __restrict__ b_hh,
                          float* __restrict__ gx)
{
    int idx = blockIdx.x * blockDim.x + threadIdx.x;
    if (idx >= TSTEPS * G4) return;
    int t = idx >> 9, g = idx & (G4 - 1);
    const float* xr = x + ((size_t)(BATCH - 1) * TSTEPS + t) * NIN;
    const float* wr = w_ih + (size_t)g * NIN;
    float acc = b_ih[g] + b_hh[g];
#pragma unroll
    for (int k = 0; k < NIN; ++k) acc += xr[k] * wr[k];
    gx[idx] = acc;
}

// ---------------------------------------------------------------------------
// Kernel 2: serial scan. One workgroup, 8 waves; wave w owns H cols [16w,16w+16)
// and computes all 4 gates for that chunk with v_wmma_f32_16x16x32_f16.
// ---------------------------------------------------------------------------
__global__ void __launch_bounds__(256, 1) lstm_scan_kernel(
    const float* __restrict__ w_hh,  // [4H, H]
    const float* __restrict__ gx,    // [T, 4H]
    float* __restrict__ hs)          // [T, H]
{
    __shared__ _Float16 hbuf[2 * 16 * HSTRIDE];   // double-buffered 16x128 h tile

    const int tid     = threadIdx.x;
    const int wv      = tid >> 5;
    const int lane    = tid & 31;
    const int half    = lane >> 4;   // 0 or 1
    const int n       = lane & 15;   // N for B/C, M for A
    const int colbase = wv * 16;

    for (int i = tid; i < 16 * HSTRIDE; i += 256) hbuf[i] = (_Float16)0.f;

    // Loop-invariant w_hh B-tiles in VGPRs (f32 -> f16).
    // B K order used here (contiguous, matches A): lanes 0-15 K=0..15, lanes 16-31 K=16..31; N = lane&15.
    v16h Bhh[4][4];   // [gate][K-chunk]
#pragma unroll
    for (int q = 0; q < 4; ++q)
#pragma unroll
        for (int kc = 0; kc < 4; ++kc) {
            const float* wr = w_hh + (size_t)(q * HDIM + colbase + n) * HDIM + kc * 32 + half * 16;
#pragma unroll
            for (int e = 0; e < 16; ++e) Bhh[q][kc][e] = (_Float16)wr[e];
        }

    int qcol[4];
#pragma unroll
    for (int q = 0; q < 4; ++q) qcol[q] = q * HDIM + colbase + n;

    v8f cst;
#pragma unroll
    for (int r = 0; r < 8; ++r) cst[r] = 0.f;

    // gx for t=0
    float gxc[4], gxn[4];
#pragma unroll
    for (int q = 0; q < 4; ++q) gxc[q] = gx[qcol[q]];

    __syncthreads();

    for (int t = 0; t < TSTEPS; ++t) {
        const _Float16* __restrict__ cur = hbuf + (t & 1) * (16 * HSTRIDE);
        _Float16* __restrict__       nxt = hbuf + ((t + 1) & 1) * (16 * HSTRIDE);

        // prefetch gx for t+1 (hidden behind WMMAs)
        if (t + 1 < TSTEPS) {
            const float* gp = gx + (size_t)(t + 1) * G4;
#pragma unroll
            for (int q = 0; q < 4; ++q) gxn[q] = gp[qcol[q]];
        }

        // A_h tiles from LDS. A K order used here (contiguous, matches B): M = lane&15,
        // element e -> K = e + half*16.
        v16h Ah[4];
#pragma unroll
        for (int kc = 0; kc < 4; ++kc)
#pragma unroll
            for (int e = 0; e < 16; ++e) {
                int K = kc * 32 + e + half * 16;
                Ah[kc][e] = cur[n * HSTRIDE + K];
            }

        // gates: C = gx (broadcast across rows) + h @ w_hh^T
        v8f Cq[4];
#pragma unroll
        for (int q = 0; q < 4; ++q) {
            v8f c;
#pragma unroll
            for (int r = 0; r < 8; ++r) c[r] = gxc[q];
#pragma unroll
            for (int kc = 0; kc < 4; ++kc)
                c = __builtin_amdgcn_wmma_f32_16x16x32_f16(false, Ah[kc], false, Bhh[q][kc],
                                                           (short)0, c, false, false);
            Cq[q] = c;
        }

        // elementwise LSTM update; C/D layout: M = r + half*8, N = lane&15
#pragma unroll
        for (int r = 0; r < 8; ++r) {
            float iv = sigf(Cq[0][r]);
            float fv = sigf(Cq[1][r]);
            float gv = tanh_fast(Cq[2][r]);
            float ov = sigf(Cq[3][r]);
            float cn = fv * cst[r] + iv * gv;
            cst[r] = cn;
            float hn = ov * tanh_fast(cn);
            int M = r + half * 8;
            nxt[M * HSTRIDE + colbase + n] = (_Float16)hn;
            if (r == 7 && half == 1)            // M==15 row (== row 255's clone)
                hs[(size_t)t * HDIM + colbase + n] = hn;
        }

#pragma unroll
        for (int q = 0; q < 4; ++q) gxc[q] = gxn[q];

        __syncthreads();
    }
}

// ---------------------------------------------------------------------------
// Kernel 3: out[t, o] = hs[t,:] . w_fc[o,:] + b_fc[o]
// ---------------------------------------------------------------------------
__global__ void fc_kernel(const float* __restrict__ hs, const float* __restrict__ w_fc,
                          const float* __restrict__ b_fc, float* __restrict__ out)
{
    int idx = blockIdx.x * blockDim.x + threadIdx.x;
    if (idx >= TSTEPS * 10) return;
    int t = idx / 10, o = idx % 10;
    const float* hr = hs + (size_t)t * HDIM;
    const float* wr = w_fc + (size_t)o * HDIM;
    float acc = b_fc[o];
#pragma unroll 8
    for (int k = 0; k < HDIM; ++k) acc += hr[k] * wr[k];
    out[idx] = acc;
}

extern "C" void kernel_launch(void* const* d_in, const int* in_sizes, int n_in,
                              void* d_out, int out_size, void* d_ws, size_t ws_size,
                              hipStream_t stream)
{
    (void)in_sizes; (void)n_in; (void)out_size; (void)ws_size;
    const float* x    = (const float*)d_in[0];
    const float* w_ih = (const float*)d_in[1];
    const float* w_hh = (const float*)d_in[2];
    const float* b_ih = (const float*)d_in[3];
    const float* b_hh = (const float*)d_in[4];
    const float* w_fc = (const float*)d_in[5];
    const float* b_fc = (const float*)d_in[6];
    float* out = (float*)d_out;

    float* hs = (float*)d_ws;                      // T*H   floats (2 MB)
    float* gxbuf = hs + (size_t)TSTEPS * HDIM;     // T*4H  floats (8 MB)

    int ngx = TSTEPS * G4;
    gx_kernel<<<(ngx + 255) / 256, 256, 0, stream>>>(x, w_ih, b_ih, b_hh, gxbuf);

    lstm_scan_kernel<<<1, 256, 0, stream>>>(w_hh, gxbuf, hs);

    int nfc = TSTEPS * 10;
    fc_kernel<<<(nfc + 255) / 256, 256, 0, stream>>>(hs, w_fc, b_fc, out);
}
